// DAWN_21732534518320
// MI455X (gfx1250) — matrix operand order repaired and checked by reading, not compiled, not run on hardware
//
#include <hip/hip_runtime.h>
#include <hip/hip_bf16.h>

// ---------------------------------------------------------------------------
// MI455X (gfx1250) implementation of the gated sense/emit transformer block.
// Dense GEMMs: v_wmma_f32_16x16x32_bf16, double-buffered LDS stages filled by
// GLOBAL_LOAD_ASYNC_TO_LDS_B128 (ASYNCcnt). Top-k gates turn the emit GEMMs
// into sparse gathers; attention is flash-style with per-wave WMMA.
// ---------------------------------------------------------------------------

typedef __attribute__((ext_vector_type(16))) __bf16 v16bf;
typedef __attribute__((ext_vector_type(8)))  float  v8f;

struct __align__(16) U16B { unsigned x[4]; };
union frag_u { U16B u[2]; v16bf v; };

constexpr int Bc  = 2;
constexpr int Sc  = 2048;
constexpr int Dc  = 1024;
constexpr int Mc  = Bc * Sc;      // 4096 tokens
constexpr int NQKc = 4096;
constexpr int NVc  = 4096;
constexpr int NKNc = 8192;
constexpr int Hc  = 16;
constexpr int DHc = 64;

// async copy: 16B from global to LDS (GV mode, no SADDR); tracked by ASYNCcnt
__device__ __forceinline__ void async_ld_b128(unsigned lds_off, const void* g) {
  asm volatile("global_load_async_to_lds_b128 %0, %1, off"
               :: "v"(lds_off), "v"(g) : "memory");
}
__device__ __forceinline__ void wait_async0() {
  asm volatile("s_wait_asynccnt 0x0" ::: "memory");
}

// ------------------------------ reductions ---------------------------------
__device__ __forceinline__ float blockReduceSum(float v, float* sred) {
  int lane = threadIdx.x & 31, wid = threadIdx.x >> 5;
#pragma unroll
  for (int o = 16; o > 0; o >>= 1) v += __shfl_xor(v, o, 32);
  if (lane == 0) sred[wid] = v;
  __syncthreads();
  int nw = (blockDim.x + 31) >> 5;
  v = (threadIdx.x < nw) ? sred[threadIdx.x] : 0.0f;
  if (wid == 0) {
#pragma unroll
    for (int o = 16; o > 0; o >>= 1) v += __shfl_xor(v, o, 32);
  }
  if (threadIdx.x == 0) sred[0] = v;
  __syncthreads();
  v = sred[0];
  __syncthreads();
  return v;
}

__device__ __forceinline__ float blockReduceMax(float v, float* sred) {
  int lane = threadIdx.x & 31, wid = threadIdx.x >> 5;
#pragma unroll
  for (int o = 16; o > 0; o >>= 1) v = fmaxf(v, __shfl_xor(v, o, 32));
  if (lane == 0) sred[wid] = v;
  __syncthreads();
  int nw = (blockDim.x + 31) >> 5;
  v = (threadIdx.x < nw) ? sred[threadIdx.x] : -3e30f;
  if (wid == 0) {
#pragma unroll
    for (int o = 16; o > 0; o >>= 1) v = fmaxf(v, __shfl_xor(v, o, 32));
  }
  if (threadIdx.x == 0) sred[0] = v;
  __syncthreads();
  v = sred[0];
  __syncthreads();
  return v;
}

// ------------------------------ small utils --------------------------------
__global__ void k_zero(float* p, size_t n) {
  size_t i = (size_t)blockIdx.x * blockDim.x + threadIdx.x;
  if (i < n) p[i] = 0.0f;
}

__global__ void k_cast_bf16(const float* __restrict__ in, __bf16* __restrict__ o, size_t n) {
  for (size_t i = (size_t)blockIdx.x * blockDim.x + threadIdx.x; i < n;
       i += (size_t)gridDim.x * blockDim.x)
    o[i] = (__bf16)in[i];
}

// o[c*R + r] = in[r*C + c]  (bf16 transpose-cast, used for expand_O)
__global__ void k_tcast_bf16(const float* __restrict__ in, __bf16* __restrict__ o,
                             int R, int C) {
  int idx = blockIdx.x * blockDim.x + threadIdx.x;
  if (idx < R * C) {
    int r = idx / C, c = idx % C;
    o[(size_t)c * R + r] = (__bf16)in[idx];
  }
}

__global__ void k_add(const float* __restrict__ a, const float* __restrict__ b,
                      float* __restrict__ o, size_t n) {
  size_t i = (size_t)blockIdx.x * blockDim.x + threadIdx.x;
  if (i < n) o[i] = a[i] + b[i];
}

__global__ void k_add_inplace(float* __restrict__ o, const float* __restrict__ a, size_t n) {
  size_t i = (size_t)blockIdx.x * blockDim.x + threadIdx.x;
  if (i < n) o[i] += a[i];
}

// ------------------------------ LayerNorm ----------------------------------
__global__ void k_layernorm(const float* __restrict__ x, const float* __restrict__ sc,
                            const float* __restrict__ bi, float* __restrict__ of,
                            __bf16* __restrict__ ob) {
  __shared__ float sred[32];
  int t = blockIdx.x;
  const float* xr = x + (size_t)t * Dc;
  float s = 0.f, s2 = 0.f;
  for (int i = threadIdx.x; i < Dc; i += blockDim.x) {
    float v = xr[i]; s += v; s2 += v * v;
  }
  float sum  = blockReduceSum(s,  sred);
  float sum2 = blockReduceSum(s2, sred);
  float mean = sum  * (1.0f / Dc);
  float var  = sum2 * (1.0f / Dc) - mean * mean;
  float rstd = rsqrtf(var + 1e-6f);
  for (int i = threadIdx.x; i < Dc; i += blockDim.x) {
    float y = (xr[i] - mean) * rstd * sc[i] + bi[i];
    of[(size_t)t * Dc + i] = y;
    ob[(size_t)t * Dc + i] = (__bf16)y;
  }
}

// --------------------- WMMA bf16 GEMM: C = A @ B^T -------------------------
// A: [M x K] bf16 row-major, B: [N x K] bf16 row-major, C: [M x N] f32.
// Block = 256 threads = 8 waves. C macro-tile 64x128; wave (wm, wn) owns a
// 32x32 sub-block = 4 WMMA accumulators (2 A-frags x 2 B-frags).
// LDS double-buffered; stages filled by global_load_async_to_lds_b128.
__global__ void k_gemm_bf16_nt(const __bf16* __restrict__ A, const __bf16* __restrict__ Bm,
                               float* __restrict__ C, int M, int N, int K) {
  __shared__ __bf16 As[2][64 * 32]  __attribute__((aligned(32)));
  __shared__ __bf16 Bs[2][128 * 32] __attribute__((aligned(32)));
  int tid  = threadIdx.x;
  int wave = tid >> 5, lane = tid & 31;
  int wm = wave & 1, wn = wave >> 1;
  int m0 = blockIdx.y * 64;
  int n0 = blockIdx.x * 128;
  int half = lane >> 4;
  int lm  = lane & 15;
  int kb  = half * 8;            // A-frag k base
  int kbB = half * 8;            // B-frag k base
  unsigned asBase = (unsigned)(size_t)(void*)&As[0][0];
  unsigned bsBase = (unsigned)(size_t)(void*)&Bs[0][0];

  // per-thread async staging: A tile 64x32 (2048 el, 8/thread),
  // B tile 128x32 (4096 el, 16/thread)
  int ia  = tid * 8;
  int ra  = ia >> 5, ca = ia & 31;
  int ib0 = tid * 8;
  int rb0 = ib0 >> 5, cb0 = ib0 & 31;
  int ib1 = tid * 8 + 2048;
  int rb1 = ib1 >> 5, cb1 = ib1 & 31;

  auto stage_load = [&](int stg, int k0) {
    async_ld_b128(asBase + (unsigned)(stg * (64 * 32) + ia) * 2u,
                  &A[(size_t)(m0 + ra) * K + k0 + ca]);
    async_ld_b128(bsBase + (unsigned)(stg * (128 * 32) + ib0) * 2u,
                  &Bm[(size_t)(n0 + rb0) * K + k0 + cb0]);
    async_ld_b128(bsBase + (unsigned)(stg * (128 * 32) + ib1) * 2u,
                  &Bm[(size_t)(n0 + rb1) * K + k0 + cb1]);
  };

  v8f acc[2][2];
#pragma unroll
  for (int mi = 0; mi < 2; ++mi)
#pragma unroll
    for (int ni = 0; ni < 2; ++ni) acc[mi][ni] = (v8f){};

  int nk = K / 32;
  stage_load(0, 0);
  wait_async0();
  __syncthreads();

  for (int kt = 0; kt < nk; ++kt) {
    int cur = kt & 1;
    if (kt + 1 < nk) stage_load(cur ^ 1, (kt + 1) * 32);

    frag_u fa[2], fb[2];
#pragma unroll
    for (int mi = 0; mi < 2; ++mi) {
      const __bf16* ap = &As[cur][(wm * 32 + mi * 16 + lm) * 32 + kb];
      fa[mi].u[0] = *(const U16B*)&ap[0];
      fa[mi].u[1] = *(const U16B*)&ap[16];
    }
#pragma unroll
    for (int ni = 0; ni < 2; ++ni) {
      const __bf16* bp = &Bs[cur][(wn * 32 + ni * 16 + lm) * 32 + kbB];
      fb[ni].u[0] = *(const U16B*)&bp[0];
      fb[ni].u[1] = *(const U16B*)&bp[16];
    }
#pragma unroll
    for (int mi = 0; mi < 2; ++mi)
#pragma unroll
      for (int ni = 0; ni < 2; ++ni)
        acc[mi][ni] = __builtin_amdgcn_wmma_f32_16x16x32_bf16(
            false, fa[mi].v, false, fb[ni].v, (short)0, acc[mi][ni], false, false);

    if (kt + 1 < nk) wait_async0();
    __syncthreads();
  }

#pragma unroll
  for (int mi = 0; mi < 2; ++mi)
#pragma unroll
    for (int ni = 0; ni < 2; ++ni)
#pragma unroll
      for (int r = 0; r < 8; ++r) {
        int m = m0 + wm * 32 + mi * 16 + r + half * 8;
        C[(size_t)m * N + n0 + wn * 32 + ni * 16 + lm] = acc[mi][ni][r];
      }
}

// ------------------------------ tau GEMV -----------------------------------
__global__ void k_tau(const float* __restrict__ normed, const float* __restrict__ Wk,
                      const float* __restrict__ bb, float* __restrict__ tauOut, int Ccols) {
  __shared__ float sred[32];
  int t = blockIdx.x;
  const float* xr = normed + (size_t)t * Dc;
  float acc[3] = {0.f, 0.f, 0.f};
  for (int i = threadIdx.x; i < Dc; i += blockDim.x) {
    float xv = xr[i];
    for (int c = 0; c < Ccols; ++c) acc[c] += xv * Wk[i * Ccols + c];
  }
  for (int c = 0; c < Ccols; ++c) {
    float r = blockReduceSum(acc[c], sred);
    if (threadIdx.x == 0) tauOut[t * Ccols + c] = r + bb[c];
  }
}

// -------------------- per-token top-k threshold gate -----------------------
__global__ void k_gate(const float* __restrict__ scores, const float* __restrict__ tau,
                       int tauStride, int tauCol, int N, int Kk,
                       int* __restrict__ oidx, float* __restrict__ oval,
                       float* __restrict__ esum) {
  extern __shared__ float srow[];
  __shared__ float sred[32];
  __shared__ int scnt;
  int t = blockIdx.x;
  const float* sc = scores + (size_t)t * N;
  float tv = tau[t * tauStride + tauCol];

  float lmax = -1e30f;
  for (int n = threadIdx.x; n < N; n += blockDim.x) {
    float raw = sc[n] - tv;
    float g   = raw > 0.0f ? raw : 1e-8f * __expf(raw);
    float eg  = __expf(g) - 1.0f;
    srow[n] = eg;
    lmax = fmaxf(lmax, eg);
  }
  __syncthreads();
  float egmax = blockReduceMax(lmax, sred);

  float lo = 0.0f, hi = egmax;
  for (int it = 0; it < 24; ++it) {
    float mid = 0.5f * (lo + hi);
    float cl = 0.0f;
    for (int n = threadIdx.x; n < N; n += blockDim.x)
      cl += (srow[n] >= mid) ? 1.0f : 0.0f;
    float cnt = blockReduceSum(cl, sred);
    if (cnt >= (float)Kk) lo = mid; else hi = mid;
  }
  float thr = lo;

  float ls = 0.0f;
  for (int n = threadIdx.x; n < N; n += blockDim.x)
    if (srow[n] >= thr) ls += srow[n];
  float gs = blockReduceSum(ls, sred) + 1e-8f;
  float strength = tanhf(egmax);

  if (threadIdx.x == 0) scnt = 0;
  __syncthreads();
  for (int n = threadIdx.x; n < N; n += blockDim.x) {
    float eg = srow[n];
    if (eg >= thr) {
      int p = atomicAdd(&scnt, 1);
      if (p < Kk) {
        float gv = eg / gs * strength;
        oidx[(size_t)t * Kk + p] = n;
        oval[(size_t)t * Kk + p] = gv * sc[n];   // gate * raw score (emit coeff)
        atomicAdd(&esum[n], gv);
      }
    }
  }
  __syncthreads();
  int c = scnt < Kk ? scnt : Kk;
  for (int p = c + threadIdx.x; p < Kk; p += blockDim.x) {
    oidx[(size_t)t * Kk + p] = 0;
    oval[(size_t)t * Kk + p] = 0.0f;
  }
}

// ------------------------- aux loss accumulation ---------------------------
__global__ void k_aux(const float* __restrict__ es, int N, float invM, float tgt,
                      float scale, float* __restrict__ outp) {
  __shared__ float sred[32];
  float l = 0.0f;
  for (int n = blockIdx.x * blockDim.x + threadIdx.x; n < N;
       n += gridDim.x * blockDim.x) {
    float d = es[n] * invM - tgt;
    l += d * d;
  }
  float s = blockReduceSum(l, sred);
  if (threadIdx.x == 0) atomicAdd(outp, s * scale);
}

// --------------------- sparse emit (gather of w rows) ----------------------
// mode 0: f32 dense [t, D]
// mode 1: bf16 head layout [b, h, s, dh]
// mode 2: bf16 transposed  [b, h, dh, s]  (for WMMA B fragments of V)
__global__ void k_emit(const int* __restrict__ idx, const float* __restrict__ val,
                       const float* __restrict__ W, int Kk, int mode,
                       float* __restrict__ of32, __bf16* __restrict__ obf) {
  int t  = blockIdx.x;
  int c0 = threadIdx.x * 4;
  float4 acc = make_float4(0.f, 0.f, 0.f, 0.f);
  for (int j = 0; j < Kk; ++j) {
    float v = val[(size_t)t * Kk + j];
    int   r = idx[(size_t)t * Kk + j];
    float4 w4 = *(const float4*)&W[(size_t)r * Dc + c0];
    acc.x += v * w4.x; acc.y += v * w4.y; acc.z += v * w4.z; acc.w += v * w4.w;
  }
  if (mode == 0) {
    *(float4*)&of32[(size_t)t * Dc + c0] = acc;
  } else {
    int b = t >> 11, s = t & (Sc - 1);
    int h = c0 >> 6, d = c0 & (DHc - 1);
    if (mode == 1) {
      size_t base = (((size_t)(b * Hc + h) * Sc) + s) * DHc + d;
      union { unsigned long long u; __bf16 b4[4]; } pk;
      pk.b4[0] = (__bf16)acc.x; pk.b4[1] = (__bf16)acc.y;
      pk.b4[2] = (__bf16)acc.z; pk.b4[3] = (__bf16)acc.w;
      *(unsigned long long*)&obf[base] = pk.u;
    } else {
      size_t bh = (size_t)(b * Hc + h) * DHc;
      obf[(bh + d + 0) * Sc + s] = (__bf16)acc.x;
      obf[(bh + d + 1) * Sc + s] = (__bf16)acc.y;
      obf[(bh + d + 2) * Sc + s] = (__bf16)acc.z;
      obf[(bh + d + 3) * Sc + s] = (__bf16)acc.w;
    }
  }
}

// --------------------- flash attention (wave per 16 queries) ---------------
// Qh/Kh: [b,h,s,dh] bf16; Vt: [b,h,dh,s] bf16; Ob: [b,s,h,dh] bf16
__global__ void k_attn(const __bf16* __restrict__ Qh, const __bf16* __restrict__ Kh,
                       const __bf16* __restrict__ Vt, __bf16* __restrict__ Ob) {
  __shared__ __bf16 Pst[4][16][32] __attribute__((aligned(32)));
  int wave = threadIdx.x >> 5;
  int lane = threadIdx.x & 31;
  int tile = blockIdx.x * 4 + wave;
  int qt = tile & (Sc / 16 - 1);
  int h  = (tile >> 7) & (Hc - 1);
  int b  = tile >> 11;
  int q0 = qt * 16;
  size_t bh = (size_t)(b * Hc + h);
  int half = lane >> 4;
  int lm   = lane & 15;
  int kb   = half * 8;

  frag_u aq[2];
  {
    const __bf16* qrow = Qh + (bh * Sc + (size_t)(q0 + lm)) * DHc;
#pragma unroll
    for (int kc = 0; kc < 2; ++kc) {
      aq[kc].u[0] = *(const U16B*)&qrow[kc * 32 + kb];
      aq[kc].u[1] = *(const U16B*)&qrow[kc * 32 + kb + 16];
    }
  }

  v8f accO[4];
#pragma unroll
  for (int c = 0; c < 4; ++c) accO[c] = (v8f){};
  float mrun[8], lrun[8];
#pragma unroll
  for (int r = 0; r < 8; ++r) { mrun[r] = -1e30f; lrun[r] = 0.0f; }

  int jmax = (q0 + 16 + 31) >> 5;
  for (int j = 0; j < jmax; ++j) {
    int key0 = j * 32;
    v8f S0 = {}, S1 = {};
#pragma unroll
    for (int nb = 0; nb < 2; ++nb) {
      v8f acc = {};
#pragma unroll
      for (int kc = 0; kc < 2; ++kc) {
        frag_u bk;
        const __bf16* krow = Kh + (bh * Sc + (size_t)(key0 + nb * 16 + lm)) * DHc
                                + kc * 32 + half * 8;
        bk.u[0] = *(const U16B*)&krow[0];
        bk.u[1] = *(const U16B*)&krow[16];
        acc = __builtin_amdgcn_wmma_f32_16x16x32_bf16(false, aq[kc].v, false, bk.v,
                                                      (short)0, acc, false, false);
      }
      if (nb == 0) S0 = acc; else S1 = acc;
    }

    float p0[8], p1[8];
#pragma unroll
    for (int r = 0; r < 8; ++r) {
      int q   = q0 + r + half * 8;
      int k0i = key0 + lm;
      int k1i = key0 + 16 + lm;
      float s0 = (k0i <= q) ? S0[r] * 0.125f : -1e30f;
      float s1 = (k1i <= q) ? S1[r] * 0.125f : -1e30f;
      float lmx = fmaxf(s0, s1);
#pragma unroll
      for (int o = 1; o < 16; o <<= 1) lmx = fmaxf(lmx, __shfl_xor(lmx, o, 16));
      float mnew  = fmaxf(mrun[r], lmx);
      float alpha = __expf(mrun[r] - mnew);
      float e0 = __expf(s0 - mnew);
      float e1 = __expf(s1 - mnew);
      float rs = e0 + e1;
#pragma unroll
      for (int o = 1; o < 16; o <<= 1) rs += __shfl_xor(rs, o, 16);
      lrun[r] = lrun[r] * alpha + rs;
      mrun[r] = mnew;
      p0[r] = e0; p1[r] = e1;
#pragma unroll
      for (int c = 0; c < 4; ++c) accO[c][r] *= alpha;
    }

#pragma unroll
    for (int r = 0; r < 8; ++r) {
      Pst[wave][r + half * 8][lm]      = (__bf16)p0[r];
      Pst[wave][r + half * 8][16 + lm] = (__bf16)p1[r];
    }
    asm volatile("s_wait_dscnt 0x0" ::: "memory");
    frag_u aP;
    aP.u[0] = *(const U16B*)&Pst[wave][lm][kb];
    aP.u[1] = *(const U16B*)&Pst[wave][lm][kb + 16];

#pragma unroll
    for (int c = 0; c < 4; ++c) {
      frag_u bV;
      const __bf16* vrow = Vt + (bh * DHc + (size_t)(c * 16 + lm)) * Sc
                              + key0 + half * 8;
      bV.u[0] = *(const U16B*)&vrow[0];
      bV.u[1] = *(const U16B*)&vrow[16];
      accO[c] = __builtin_amdgcn_wmma_f32_16x16x32_bf16(false, aP.v, false, bV.v,
                                                        (short)0, accO[c], false, false);
    }
  }

#pragma unroll
  for (int r = 0; r < 8; ++r) {
    int q = q0 + r + half * 8;
    float inv = 1.0f / lrun[r];
    size_t base = (((size_t)b * Sc + q) * Hc + h) * DHc;
#pragma unroll
    for (int c = 0; c < 4; ++c)
      Ob[base + c * 16 + lm] = (__bf16)(accO[c][r] * inv);
  }
}

// ===========================================================================
extern "C" void kernel_launch(void* const* d_in, const int* in_sizes, int n_in,
                              void* d_out, int out_size, void* d_ws, size_t ws_size,
                              hipStream_t stream) {
  (void)in_sizes; (void)n_in; (void)out_size; (void)ws_size;
  const float* x          = (const float*)d_in[0];
  const float* qk_emb     = (const float*)d_in[1];
  const float* qk_w       = (const float*)d_in[2];
  const float* v_emb      = (const float*)d_in[3];
  const float* v_w        = (const float*)d_in[4];
  const float* know_emb   = (const float*)d_in[5];
  const float* know_w     = (const float*)d_in[6];
  const float* tau_attn_k = (const float*)d_in[7];
  const float* tau_attn_b = (const float*)d_in[8];
  const float* tau_know_k = (const float*)d_in[9];
  const float* tau_know_b = (const float*)d_in[10];
  const float* expand_O   = (const float*)d_in[11];
  const float* ln1_scale  = (const float*)d_in[12];
  const float* ln1_bias   = (const float*)d_in[13];
  const float* ln2_scale  = (const float*)d_in[14];
  const float* ln2_bias   = (const float*)d_in[15];
  float* out = (float*)d_out;

  char* w = (char*)d_ws;
  auto carve = [&](size_t bytes) -> void* {
    void* p = (void*)w;
    w += (bytes + 255) & ~(size_t)255;
    return p;
  };
  float*  normedf = (float*) carve((size_t)Mc * Dc * 4);
  __bf16* normedb = (__bf16*)carve((size_t)Mc * Dc * 2);
  __bf16* wbuf    = (__bf16*)carve((size_t)NKNc * Dc * 2);
  float*  scores  = (float*) carve((size_t)Mc * NKNc * 4);
  float*  taub    = (float*) carve((size_t)Mc * 3 * 4);
  int*    idxQ    = (int*)   carve((size_t)Mc * 32 * 4);
  float*  valQ    = (float*) carve((size_t)Mc * 32 * 4);
  int*    idxK    = (int*)   carve((size_t)Mc * 32 * 4);
  float*  valK    = (float*) carve((size_t)Mc * 32 * 4);
  int*    idxV    = (int*)   carve((size_t)Mc * 32 * 4);
  float*  valV    = (float*) carve((size_t)Mc * 32 * 4);
  int*    idxN    = (int*)   carve((size_t)Mc * 64 * 4);
  float*  valN    = (float*) carve((size_t)Mc * 64 * 4);
  float*  esumAll = (float*) carve((size_t)(3 * NQKc + NKNc) * 4);
  float*  esumQ = esumAll, *esumK = esumAll + NQKc, *esumV = esumAll + 2 * NQKc;
  float*  esumN = esumAll + 3 * NQKc;
  __bf16* Qh   = (__bf16*)carve((size_t)Mc * Dc * 2);
  __bf16* Kh   = (__bf16*)carve((size_t)Mc * Dc * 2);
  __bf16* Vt   = (__bf16*)carve((size_t)Mc * Dc * 2);
  __bf16* attO = (__bf16*)carve((size_t)Mc * Dc * 2);
  float*  proj = (float*) carve((size_t)Mc * Dc * 4);
  float*  xmid = (float*) carve((size_t)Mc * Dc * 4);

  const size_t nMD = (size_t)Mc * Dc;
  float* auxp = out + nMD;

  k_zero<<<1, 1, 0, stream>>>(auxp, 1);
  k_zero<<<(3 * NQKc + NKNc + 255) / 256, 256, 0, stream>>>(esumAll, 3 * NQKc + NKNc);

  // ---- LN1 ----
  k_layernorm<<<Mc, 256, 0, stream>>>(x, ln1_scale, ln1_bias, normedf, normedb);

  // ---- scores_qk, gates Q/K ----
  k_cast_bf16<<<4096, 256, 0, stream>>>(qk_emb, wbuf, (size_t)NQKc * Dc);
  k_gemm_bf16_nt<<<dim3(NQKc / 128, Mc / 64), 256, 0, stream>>>(
      normedb, wbuf, scores, Mc, NQKc, Dc);
  k_tau<<<Mc, 128, 0, stream>>>(normedf, tau_attn_k, tau_attn_b, taub, 3);
  k_gate<<<Mc, 256, NQKc * 4, stream>>>(scores, taub, 3, 0, NQKc, 32, idxQ, valQ, esumQ);
  k_gate<<<Mc, 256, NQKc * 4, stream>>>(scores, taub, 3, 1, NQKc, 32, idxK, valK, esumK);

  // ---- scores_v (reuse buffer), gate V ----
  k_cast_bf16<<<4096, 256, 0, stream>>>(v_emb, wbuf, (size_t)NVc * Dc);
  k_gemm_bf16_nt<<<dim3(NVc / 128, Mc / 64), 256, 0, stream>>>(
      normedb, wbuf, scores, Mc, NVc, Dc);
  k_gate<<<Mc, 256, NVc * 4, stream>>>(scores, taub, 3, 2, NVc, 32, idxV, valV, esumV);

  // ---- aux (attention part) ----
  k_aux<<<8, 256, 0, stream>>>(esumQ, NQKc, 1.0f / Mc, 1.0f / NQKc, (float)NQKc, auxp);
  k_aux<<<8, 256, 0, stream>>>(esumK, NQKc, 1.0f / Mc, 1.0f / NQKc, (float)NQKc, auxp);
  k_aux<<<8, 256, 0, stream>>>(esumV, NVc,  1.0f / Mc, 1.0f / NVc,  (float)NVc,  auxp);

  // ---- sparse emits ----
  k_emit<<<Mc, 256, 0, stream>>>(idxQ, valQ, qk_w, 32, 1, nullptr, Qh);
  k_emit<<<Mc, 256, 0, stream>>>(idxK, valK, qk_w, 32, 1, nullptr, Kh);
  k_emit<<<Mc, 256, 0, stream>>>(idxV, valV, v_w,  32, 2, nullptr, Vt);

  // ---- flash attention ----
  k_attn<<<(Bc * Hc * (Sc / 16)) / 4, 128, 0, stream>>>(Qh, Kh, Vt, attO);

  // ---- out @ expand_O, residual ----
  k_tcast_bf16<<<(Dc * Dc + 255) / 256, 256, 0, stream>>>(expand_O, wbuf, Dc, Dc);
  k_gemm_bf16_nt<<<dim3(Dc / 128, Mc / 64), 256, 0, stream>>>(
      attO, wbuf, proj, Mc, Dc, Dc);
  k_add<<<(int)((nMD + 255) / 256), 256, 0, stream>>>(x, proj, xmid, nMD);

  // ---- LN2, knowledge path ----
  k_layernorm<<<Mc, 256, 0, stream>>>(xmid, ln2_scale, ln2_bias, normedf, normedb);
  k_cast_bf16<<<8192, 256, 0, stream>>>(know_emb, wbuf, (size_t)NKNc * Dc);
  k_gemm_bf16_nt<<<dim3(NKNc / 128, Mc / 64), 256, 0, stream>>>(
      normedb, wbuf, scores, Mc, NKNc, Dc);
  k_tau<<<Mc, 128, 0, stream>>>(normedf, tau_know_k, tau_know_b, taub, 1);
  k_gate<<<Mc, 256, NKNc * 4, stream>>>(scores, taub, 1, 0, NKNc, 64, idxN, valN, esumN);
  k_aux<<<8, 256, 0, stream>>>(esumN, NKNc, 1.0f / Mc, 1.0f / NKNc, (float)NKNc, auxp);

  k_emit<<<Mc, 256, 0, stream>>>(idxN, valN, know_w, 64, 0, out, nullptr);
  k_add_inplace<<<(int)((nMD + 255) / 256), 256, 0, stream>>>(out, xmid, nMD);
}
